// Upsampling_11562051960841
// MI455X (gfx1250) — matrix operand order repaired and checked by reading, not compiled, not run on hardware
//
#include <hip/hip_runtime.h>
#include <hip/hip_bf16.h>

// ---------------------------------------------------------------------------
// CDNA5 (gfx1250) deformable-conv + conv-transpose pipeline, f16 WMMA GEMMs,
// double-buffered LDS B-tiles (one barrier per K-chunk, gather/WMMA overlap).
// ---------------------------------------------------------------------------

typedef __attribute__((ext_vector_type(16))) _Float16 v16h;
typedef __attribute__((ext_vector_type(8)))  float    v8f;

union AF { v16h v; uint4 q[2]; };     // 32B WMMA A/B fragment
union CF { v8f  v; float f[8]; };     // 32B WMMA C/D accumulator
union PK2 { _Float16 h[4]; uint2 u; };

#define KTOT1 2304   // 9 taps * 256 ch  (deformable GEMM K)
#define KTOT2 1024   // 4 taps * 256 ch  (convT GEMM K)

// ---------------------------------------------------------------------------
// Prep: repack w_reg (Co,C,3,3) -> Areg[o][k*256+c] f16
// ---------------------------------------------------------------------------
__global__ __launch_bounds__(256) void k_prep_areg(const float* __restrict__ w_reg,
                                                   _Float16* __restrict__ Areg) {
  int i = blockIdx.x * 256 + threadIdx.x;           // < 256*2304
  if (i >= 256 * KTOT1) return;
  int o = i / KTOT1, r = i % KTOT1;
  int k = r >> 8, c = r & 255;
  Areg[i] = (_Float16)w_reg[(o * 256 + c) * 9 + k];
}

// ---------------------------------------------------------------------------
// Prep: repack w_up (C,Co,4,4) -> 4 parity GEMM matrices Aup[p][o][t*256+c]
// parity p = py*2+px of the OUTPUT pixel; tap t = tyi*2+txi selects the 2x2
// kernel taps that contribute to that parity (stride-2, pad-1, k=4).
// ---------------------------------------------------------------------------
__global__ __launch_bounds__(256) void k_prep_aup(const float* __restrict__ w_up,
                                                  _Float16* __restrict__ Aup) {
  int i = blockIdx.x * 256 + threadIdx.x;           // exactly 4*256*1024
  int p = i >> 18;
  int rem = i & 262143;
  int o = rem >> 10;
  int r = rem & 1023;
  int t = r >> 8, c = r & 255;
  int tyi = t >> 1, txi = t & 1;
  int pyp = p >> 1, pxp = p & 1;
  int ky = pyp ? (tyi ? 2 : 0) : (tyi ? 3 : 1);
  int kx = pxp ? (txi ? 2 : 0) : (txi ? 3 : 1);
  Aup[i] = (_Float16)w_up[((c * 256 + o) * 4 + ky) * 4 + kx];
}

// ---------------------------------------------------------------------------
// Prep: fold BN params -> scale/bias   SB[0:256]=s1 [256:512]=b1 [512:768]=s2
// [768:1024]=b2
// ---------------------------------------------------------------------------
__global__ void k_prep_bn(const float* g1, const float* b1, const float* m1, const float* v1,
                          const float* g2, const float* b2, const float* m2, const float* v2,
                          float* SB) {
  int i = threadIdx.x;
  float s1 = g1[i] * rsqrtf(v1[i] + 1e-5f);
  SB[i]       = s1;
  SB[256 + i] = b1[i] - m1[i] * s1;
  float s2 = g2[i] * rsqrtf(v2[i] + 1e-5f);
  SB[512 + i] = s2;
  SB[768 + i] = b2[i] - m2[i] * s2;
}

// ---------------------------------------------------------------------------
// Offset + mask 3x3 convs (27 out ch) -> absolute sample coords + mask.
// 128 blocks: 8 images x 16 spatial 16x16 tiles. Direct VALU conv (small).
// ---------------------------------------------------------------------------
__global__ __launch_bounds__(256) void k_offmask(
    const float* __restrict__ x, const float* __restrict__ w_off,
    const float* __restrict__ b_off, const float* __restrict__ w_mod,
    const float* __restrict__ b_mod, float* __restrict__ PY,
    float* __restrict__ PX, float* __restrict__ PM) {
  __shared__ float xs[8][18][19];
  __shared__ float wts[1944];          // [c][tap][oc] for the 8-ch chunk
  int tid = threadIdx.x;
  int b = blockIdx.x >> 4;
  int tileid = blockIdx.x & 15;
  int y0 = (tileid >> 2) * 16, x0 = (tileid & 3) * 16;
  int yl = tid >> 4, xl = tid & 15;

  float acc[27];
#pragma unroll
  for (int i = 0; i < 27; ++i) acc[i] = 0.f;

  for (int cb = 0; cb < 256; cb += 8) {
    for (int i = tid; i < 8 * 324; i += 256) {
      int c = i / 324, pos = i % 324;
      int yy = pos / 18, xx = pos % 18;
      int gy = y0 - 1 + yy, gx = x0 - 1 + xx;
      float v = 0.f;
      if (gy >= 0 && gy < 64 && gx >= 0 && gx < 64)
        v = x[((b * 256 + cb + c) << 12) + gy * 64 + gx];
      xs[c][yy][xx] = v;
    }
    for (int i = tid; i < 1944; i += 256) {
      int c = i / 243, rem = i % 243, tap = rem / 27, oc = rem % 27;
      float wv;
      if (oc < 18) wv = w_off[(oc * 256 + cb + c) * 9 + tap];
      else         wv = w_mod[((oc - 18) * 256 + cb + c) * 9 + tap];
      wts[(c * 9 + tap) * 27 + oc] = wv;
    }
    __syncthreads();
    for (int c = 0; c < 8; ++c) {
      for (int tap = 0; tap < 9; ++tap) {
        float xv = xs[c][yl + tap / 3][xl + tap % 3];
        const float* wr = &wts[(c * 9 + tap) * 27];
#pragma unroll
        for (int oc = 0; oc < 27; ++oc) acc[oc] += xv * wr[oc];
      }
    }
    __syncthreads();
  }

  int gy = y0 + yl, gx = x0 + xl;
  int hw = gy * 64 + gx;
#pragma unroll
  for (int k = 0; k < 9; ++k) {
    float oy = acc[2 * k]     + b_off[2 * k];
    float ox = acc[2 * k + 1] + b_off[2 * k + 1];
    float z  = acc[18 + k]    + b_mod[k];
    float m  = 2.f / (1.f + __expf(-z));
    int idx = (b * 9 + k) * 4096 + hw;
    PY[idx] = (float)gy - 1.f + (float)(k / 3) + oy;
    PX[idx] = (float)gx - 1.f + (float)(k % 3) + ox;
    PM[idx] = m;
  }
}

// ---------------------------------------------------------------------------
// Deformable GEMM: O[256 x 32768] = Areg[256 x 2304] * V[2304 x 32768],
// V sampled on the fly into a double-buffered LDS B-tile (32 px x 32 K, f16).
// K ordered tap-major so a 32-wide chunk shares one bilinear footprint;
// all 9 taps' corner weights/indices precomputed into LDS. 1024 blocks x
// 256 threads (8 waves); each wave: 32x32 output sub-tile, 4x
// v_wmma_f32_16x16x32_f16 per K-chunk, one barrier per chunk (fill of chunk
// q+1 overlaps WMMA of chunk q). Epilogue: BN1+ReLU -> HPK pixel-major f16.
// ---------------------------------------------------------------------------
__global__ __launch_bounds__(256) void k_dconv(
    const float* __restrict__ x, const _Float16* __restrict__ Areg,
    const float* __restrict__ PY, const float* __restrict__ PX,
    const float* __restrict__ PM, const float* __restrict__ SB,
    _Float16* __restrict__ HPK) {
  __shared__ __align__(16) _Float16 Bt[2][32][40];  // 80B stride: conflict-free
  __shared__ float cw[9][4][32];
  __shared__ int   cidx[9][4][32];

  int tid = threadIdx.x;
  int wv = tid >> 5, lane = tid & 31;
  int gbase = blockIdx.x * 32;
  int fp = tid >> 3, fc = (tid & 7) * 4;            // B-fill: pixel / channel

  // Bilinear corner setup for all 9 taps x 32 pixels (288 items).
  for (int it = tid; it < 288; it += 256) {
    int k = it >> 5, p = it & 31;
    int g = gbase + p;
    int b = g >> 12, hw = g & 4095;
    int base = (b * 9 + k) * 4096 + hw;
    float py = PY[base], px = PX[base], m = PM[base];
    float y0f = floorf(py), x0f = floorf(px);
    float wy1 = py - y0f, wy0 = 1.f - wy1;
    float wx1 = px - x0f, wx0 = 1.f - wx1;
    int yi0 = (int)y0f, xi0 = (int)x0f;
    int xb = b << 20;                               // b*256*4096 (channel 0)
#pragma unroll
    for (int j = 0; j < 4; ++j) {
      int yi = yi0 + (j >> 1), xi = xi0 + (j & 1);
      float wgt = ((j >> 1) ? wy1 : wy0) * ((j & 1) ? wx1 : wx0) * m;
      bool okc = (yi >= 0) && (yi < 64) && (xi >= 0) && (xi < 64);
      int yc = min(max(yi, 0), 63), xc = min(max(xi, 0), 63);
      cw[k][j][p] = okc ? wgt : 0.f;
      cidx[k][j][p] = xb + yc * 64 + xc;
    }
  }

  v8f zero = {};
  CF acc[2][2];
#pragma unroll
  for (int mi = 0; mi < 2; ++mi)
#pragma unroll
    for (int ni = 0; ni < 2; ++ni) acc[mi][ni].v = zero;

  auto fill = [&](int q, int buf) {
    int k = q >> 3;
    int ccfc = ((q & 7) << 5) + fc;
    float w0 = cw[k][0][fp], w1 = cw[k][1][fp];
    float w2 = cw[k][2][fp], w3 = cw[k][3][fp];
    int i0 = cidx[k][0][fp], i1 = cidx[k][1][fp];
    int i2 = cidx[k][2][fp], i3 = cidx[k][3][fp];
    PK2 pk;
#pragma unroll
    for (int j = 0; j < 4; ++j) {
      int c4 = (ccfc + j) << 12;
      float v = w0 * x[i0 + c4] + w1 * x[i1 + c4] +
                w2 * x[i2 + c4] + w3 * x[i3 + c4];
      pk.h[j] = (_Float16)v;
    }
    *(uint2*)&Bt[buf][fp][fc] = pk.u;
  };

  __syncthreads();                                  // corner data visible
  fill(0, 0);
  __syncthreads();

  for (int q = 0; q < 72; ++q) {                    // 72 = 2304/32 K-chunks
    if (q + 1 < 72) fill(q + 1, (q + 1) & 1);       // overlaps WMMA below

    int sel = (lane >= 16) ? 16 : 0;
    const char* ar0 =
        (const char*)Areg + (32 * wv + (lane & 15)) * (KTOT1 * 2) + q * 64;
    AF a[2];
    a[0].q[0] = *(const uint4*)(ar0 + sel);
    a[0].q[1] = *(const uint4*)(ar0 + 32 + sel);
    const char* ar1 = ar0 + 16 * (KTOT1 * 2);
    a[1].q[0] = *(const uint4*)(ar1 + sel);
    a[1].q[1] = *(const uint4*)(ar1 + 32 + sel);

    int bsel = (lane >= 16) ? 16 : 0;
    AF bf[2];
#pragma unroll
    for (int ni = 0; ni < 2; ++ni) {
      const char* bp = (const char*)&Bt[q & 1][16 * ni + (lane & 15)][0] + bsel;
      bf[ni].q[0] = *(const uint4*)bp;
      bf[ni].q[1] = *(const uint4*)(bp + 32);
    }
#pragma unroll
    for (int mi = 0; mi < 2; ++mi)
#pragma unroll
      for (int ni = 0; ni < 2; ++ni)
        acc[mi][ni].v = __builtin_amdgcn_wmma_f32_16x16x32_f16(
            false, a[mi].v, false, bf[ni].v, (short)0, acc[mi][ni].v,
            false, false);
    __syncthreads();
  }

  // BN1 + ReLU -> HPK[pixel][channel] f16
  const float* sc = SB;
  const float* bi = SB + 256;
  int lhalf = (lane >= 16) ? 8 : 0;
#pragma unroll
  for (int mi = 0; mi < 2; ++mi)
#pragma unroll
    for (int r = 0; r < 8; ++r) {
      int row = 32 * wv + 16 * mi + r + lhalf;
      float s = sc[row], o = bi[row];
#pragma unroll
      for (int ni = 0; ni < 2; ++ni) {
        int pix = gbase + 16 * ni + (lane & 15);
        float val = fmaxf(fmaf(acc[mi][ni].f[r], s, o), 0.f);
        HPK[pix * 256 + row] = (_Float16)val;
      }
    }
}

// ---------------------------------------------------------------------------
// Transposed conv as 4 parity GEMMs: O_p[256 x 32768] = Aup[p] * H_p,
// K = tap*256 + c (4 taps of the 2x2 effective kernel per parity).
// 4096 blocks (4 parity x 1024 pixel tiles) x 256 threads, double-buffered
// B-tile, one barrier per chunk. Epilogue: BN2+ReLU -> f32 (8,256,128,128).
// ---------------------------------------------------------------------------
__global__ __launch_bounds__(256) void k_convt(
    const _Float16* __restrict__ HPK, const _Float16* __restrict__ Aup,
    const float* __restrict__ SB, float* __restrict__ out) {
  __shared__ __align__(16) _Float16 Bt[2][32][40];
  int tid = threadIdx.x;
  int wv = tid >> 5, lane = tid & 31;
  int pcl = blockIdx.x >> 10;
  int tile = blockIdx.x & 1023;
  int pyp = pcl >> 1, pxp = pcl & 1;
  int gbase = tile * 32;

  const _Float16* A = Aup + (size_t)pcl * 256 * KTOT2;

  int fp = tid >> 3, fc = (tid & 7) * 4;
  int g = gbase + fp;
  int b = g >> 12, hw = g & 4095;
  int y = hw >> 6, xq = hw & 63;

  v8f zero = {};
  CF acc[2][2];
#pragma unroll
  for (int mi = 0; mi < 2; ++mi)
#pragma unroll
    for (int ni = 0; ni < 2; ++ni) acc[mi][ni].v = zero;

  auto fill = [&](int q, int buf) {
    int t = q >> 3;
    int cc = (q & 7) << 5;
    int tyi = t >> 1, txi = t & 1;
    int dy = pyp ? (tyi ? 0 : 1) : (tyi ? -1 : 0);
    int dx = pxp ? (txi ? 0 : 1) : (txi ? -1 : 0);
    int iy = y + dy, ix = xq + dx;
    bool okp = (iy >= 0) && (iy < 64) && (ix >= 0) && (ix < 64);
    PK2 pk;
    pk.u.x = 0u; pk.u.y = 0u;
    if (okp)
      pk.u = *(const uint2*)(HPK + (((b << 12) + iy * 64 + ix) * 256 + cc + fc));
    *(uint2*)&Bt[buf][fp][fc] = pk.u;
  };

  fill(0, 0);
  __syncthreads();

  for (int q = 0; q < 32; ++q) {                    // 32 = 1024/32 K-chunks
    if (q + 1 < 32) fill(q + 1, (q + 1) & 1);       // overlaps WMMA below

    int sel = (lane >= 16) ? 16 : 0;
    const char* ar0 =
        (const char*)A + (32 * wv + (lane & 15)) * (KTOT2 * 2) + q * 64;
    AF a[2];
    a[0].q[0] = *(const uint4*)(ar0 + sel);
    a[0].q[1] = *(const uint4*)(ar0 + 32 + sel);
    const char* ar1 = ar0 + 16 * (KTOT2 * 2);
    a[1].q[0] = *(const uint4*)(ar1 + sel);
    a[1].q[1] = *(const uint4*)(ar1 + 32 + sel);

    int bsel = (lane >= 16) ? 16 : 0;
    AF bf[2];
#pragma unroll
    for (int ni = 0; ni < 2; ++ni) {
      const char* bp = (const char*)&Bt[q & 1][16 * ni + (lane & 15)][0] + bsel;
      bf[ni].q[0] = *(const uint4*)bp;
      bf[ni].q[1] = *(const uint4*)(bp + 32);
    }
#pragma unroll
    for (int mi = 0; mi < 2; ++mi)
#pragma unroll
      for (int ni = 0; ni < 2; ++ni)
        acc[mi][ni].v = __builtin_amdgcn_wmma_f32_16x16x32_f16(
            false, a[mi].v, false, bf[ni].v, (short)0, acc[mi][ni].v,
            false, false);
    __syncthreads();
  }

  // BN2 + ReLU -> out f32
  const float* sc = SB + 512;
  const float* bi = SB + 768;
  int lhalf = (lane >= 16) ? 8 : 0;
#pragma unroll
  for (int mi = 0; mi < 2; ++mi)
#pragma unroll
    for (int r = 0; r < 8; ++r) {
      int row = 32 * wv + 16 * mi + r + lhalf;
      float s = sc[row], o = bi[row];
#pragma unroll
      for (int ni = 0; ni < 2; ++ni) {
        int pix = gbase + 16 * ni + (lane & 15);
        int bo = pix >> 12, hwo = pix & 4095;
        int oy = 2 * (hwo >> 6) + pyp;
        int ox = 2 * (hwo & 63) + pxp;
        float val = fmaxf(fmaf(acc[mi][ni].f[r], s, o), 0.f);
        out[((bo * 256 + row) << 14) + (oy << 7) + ox] = val;
      }
    }
}

// ---------------------------------------------------------------------------
// Host launcher. Workspace layout (bytes):
//   0        Areg  f16 256x2304  (1,179,648)
//   1179648  Aup   f16 4x256x1024 (2,097,152)
//   3276800  PY    f32 8x9x4096  (1,179,648)
//   4456448  PX    f32           (1,179,648)
//   5636096  PM    f32           (1,179,648)
//   6815744  SB    f32 4x256     (4,096)
//   6819840  HPK   f16 8x4096x256 (16,777,216)   total ~23.6 MB
// ---------------------------------------------------------------------------
extern "C" void kernel_launch(void* const* d_in, const int* in_sizes, int n_in,
                              void* d_out, int out_size, void* d_ws,
                              size_t ws_size, hipStream_t stream) {
  (void)in_sizes; (void)n_in; (void)out_size; (void)ws_size;
  const float* x     = (const float*)d_in[0];
  const float* w_off = (const float*)d_in[1];
  const float* b_off = (const float*)d_in[2];
  const float* w_mod = (const float*)d_in[3];
  const float* b_mod = (const float*)d_in[4];
  const float* w_reg = (const float*)d_in[5];
  const float* g1    = (const float*)d_in[6];
  const float* be1   = (const float*)d_in[7];
  const float* m1    = (const float*)d_in[8];
  const float* v1    = (const float*)d_in[9];
  const float* w_up  = (const float*)d_in[10];
  const float* g2    = (const float*)d_in[11];
  const float* be2   = (const float*)d_in[12];
  const float* m2    = (const float*)d_in[13];
  const float* v2    = (const float*)d_in[14];

  char* ws = (char*)d_ws;
  _Float16* Areg = (_Float16*)(ws + 0);
  _Float16* Aup  = (_Float16*)(ws + 1179648);
  float* PY = (float*)(ws + 3276800);
  float* PX = (float*)(ws + 4456448);
  float* PM = (float*)(ws + 5636096);
  float* SB = (float*)(ws + 6815744);
  _Float16* HPK = (_Float16*)(ws + 6819840);

  k_prep_areg<<<2304, 256, 0, stream>>>(w_reg, Areg);
  k_prep_aup<<<4096, 256, 0, stream>>>(w_up, Aup);
  k_prep_bn<<<1, 256, 0, stream>>>(g1, be1, m1, v1, g2, be2, m2, v2, SB);
  k_offmask<<<128, 256, 0, stream>>>(x, w_off, b_off, w_mod, b_mod, PY, PX, PM);
  k_dconv<<<1024, 256, 0, stream>>>(x, Areg, PY, PX, PM, SB, HPK);
  k_convt<<<4096, 256, 0, stream>>>(HPK, Aup, SB, (float*)d_out);
}
